// GPT2RSMBlock_53266184405092
// MI455X (gfx1250) — matrix operand order repaired and checked by reading, not compiled, not run on hardware
//
#include <hip/hip_runtime.h>
#include <math.h>

// ---------------- types / helpers ----------------
typedef __attribute__((ext_vector_type(16))) __bf16 v16bf;
typedef __attribute__((ext_vector_type(8)))  float  v8f;

static constexpr int BB = 1024;   // batch
static constexpr int DD = 1024;   // n_embd
static constexpr int SS = 64;     // conv slots
static constexpr int FF = 4096;   // ffn dim

__device__ __forceinline__ __bf16 f2bf(float f) {
  union { float f; unsigned u; } v; v.f = f;
  unsigned r = (v.u + 0x7FFFu + ((v.u >> 16) & 1u)) >> 16;   // RNE
  union { unsigned short s; __bf16 b; } o; o.s = (unsigned short)r;
  return o.b;
}

__device__ __forceinline__ float gelu_tanh(float v) {
  float c = v + 0.044715f * v * v * v;
  return 0.5f * v * (1.0f + tanhf(0.7978845608028654f * c));
}

// ---------------- weight pre-conversion kernels ----------------
// dst[c][r] = bf16(src[r][c])  -- convert + transpose, 32x32 LDS tiles
__global__ __launch_bounds__(256)
void convT_kernel(const float* __restrict__ src, __bf16* __restrict__ dst, int R, int C)
{
  __shared__ float t[32][33];
  const int r0 = blockIdx.y * 32, c0 = blockIdx.x * 32;
  const int col = threadIdx.x & 31, rw = threadIdx.x >> 5;
#pragma unroll
  for (int i = 0; i < 4; ++i) {
    const int r = rw + i * 8;
    t[r][col] = src[(size_t)(r0 + r) * C + c0 + col];
  }
  __syncthreads();
#pragma unroll
  for (int i = 0; i < 4; ++i) {
    const int oc = rw + i * 8;
    dst[(size_t)(c0 + oc) * R + r0 + col] = f2bf(t[col][oc]);
  }
}

// straight convert (layout preserved)
__global__ __launch_bounds__(256)
void conv_kernel(const float* __restrict__ src, __bf16* __restrict__ dst, int n)
{
  for (int i = blockIdx.x * 1024 + threadIdx.x; i < n; i += gridDim.x * 1024)
  {
    dst[i] = f2bf(src[i]);
    int i2 = i + 256; if (i2 < n) dst[i2] = f2bf(src[i2]);
    int i3 = i + 512; if (i3 < n) dst[i3] = f2bf(src[i3]);
    int i4 = i + 768; if (i4 < n) dst[i4] = f2bf(src[i4]);
  }
}

// ---------------- WMMA GEMM (async double-buffered B) ----------------
// C[M,N] = epilogue( A_f32[M,K] @ Bt_bf16[N,ldb]^T + bias )  (+ resid)
// Bt is bf16, row n holds the K-contiguous weights for output column n.
// Block tile 128x128, 8 waves of 32x64 (2x4 WMMA tiles), K-step 32.
// B tiles: global_load_async_to_lds_b128 (ASYNCcnt), double-buffered.
template <int GELU, int RESID, int BIAS>
__global__ __launch_bounds__(256)
void wmma_gemm(const float* __restrict__ A, const __bf16* __restrict__ Bt,
               const float* __restrict__ bias, const float* __restrict__ resid,
               float* __restrict__ C, int M, int N, int K, int ldb)
{
  constexpr int BK = 32;
  __shared__ __align__(16) __bf16 As[2][128][BK + 8];   // [buf][row][k]
  __shared__ __align__(16) __bf16 Bs[2][128][BK + 8];   // [buf][col][k]

  const int tid  = threadIdx.x;
  const int wave = tid >> 5;
  const int lane = tid & 31;
  const int half = lane >> 4;
  const int l16  = lane & 15;
  const int wm   = (wave & 3) * 32;    // wave row offset in block tile
  const int wn   = (wave >> 2) * 64;   // wave col offset in block tile
  const int m0   = blockIdx.y * 128;
  const int n0   = blockIdx.x * 128;

  // stage A tile (128x32 fp32 -> bf16), coalesced float4 + VALU convert
  auto stageA = [&](int kt, int buf) {
    const int k0 = kt * BK;
#pragma unroll
    for (int i = 0; i < 4; ++i) {
      int f   = tid + i * 256;         // 1024 float4 slots, 8 per row
      int row = f >> 3;
      int c4  = (f & 7) << 2;
      const float4 v = *(const float4*)(A + (size_t)(m0 + row) * K + k0 + c4);
      As[buf][row][c4 + 0] = f2bf(v.x); As[buf][row][c4 + 1] = f2bf(v.y);
      As[buf][row][c4 + 2] = f2bf(v.z); As[buf][row][c4 + 3] = f2bf(v.w);
    }
  };
  // stage B tile via async global->LDS DMA: 512 x 16B chunks, 2 per thread
  auto stageB = [&](int kt, int buf) {
    const int k0 = kt * BK;
#pragma unroll
    for (int i = 0; i < 2; ++i) {
      int cid = tid * 2 + i;           // 0..511
      int row = cid >> 2;              // tile col (output column)
      int ch  = cid & 3;               // 16B chunk within the 64B row
      const __bf16* g = Bt + (size_t)(n0 + row) * ldb + k0 + ch * 8;
      unsigned lds = (unsigned)(size_t)&Bs[buf][row][ch * 8];
      unsigned long long ga = (unsigned long long)(size_t)g;
      asm volatile("global_load_async_to_lds_b128 %0, %1, off"
                   :: "v"(lds), "v"(ga) : "memory");
    }
  };

  v8f acc[2][4];
#pragma unroll
  for (int i = 0; i < 2; ++i)
#pragma unroll
    for (int j = 0; j < 4; ++j) acc[i][j] = 0.0f;

  const int nk = K >> 5;
  stageB(0, 0);
  stageA(0, 0);
  for (int kt = 0; kt < nk; ++kt) {
    const int cur = kt & 1;
    if (kt + 1 < nk) {
      stageB(kt + 1, cur ^ 1);                       // overlap next-tile DMA
      stageA(kt + 1, cur ^ 1);
      __builtin_prefetch(A + (size_t)(m0 + (tid >> 1)) * K + (kt + 2) * BK, 0, 0);
      asm volatile("s_wait_asynccnt 0x2" ::: "memory");  // tile kt's copies done
    } else {
      asm volatile("s_wait_asynccnt 0x0" ::: "memory");
    }
    __syncthreads();                                 // cross-wave visibility

    // fragments: A row / B col = l16, contiguous k = 16*half + e in both
    v16bf af[2], bfr[4];
#pragma unroll
    for (int i = 0; i < 2; ++i) {
      const __bf16* row = &As[cur][wm + i * 16 + l16][0];
#pragma unroll
      for (int e = 0; e < 16; ++e) {
        int kk = half * 16 + e;  // A: K=half*16+e
        af[i][e] = row[kk];
      }
    }
#pragma unroll
    for (int j = 0; j < 4; ++j) {
      const __bf16* col = &Bs[cur][wn + j * 16 + l16][half * 16]; // B: K=half*16+e
#pragma unroll
      for (int e = 0; e < 16; ++e) bfr[j][e] = col[e];
    }
#pragma unroll
    for (int i = 0; i < 2; ++i)
#pragma unroll
      for (int j = 0; j < 4; ++j)
        acc[i][j] = __builtin_amdgcn_wmma_f32_16x16x32_bf16(
            false, af[i], false, bfr[j], (short)0, acc[i][j], false, false);
    __syncthreads();                                 // before reusing buffer `cur`
  }

  // epilogue: C layout row M = v + 8*half, col N = l16
#pragma unroll
  for (int i = 0; i < 2; ++i)
#pragma unroll
    for (int j = 0; j < 4; ++j) {
      const int col = n0 + wn + j * 16 + l16;
      const float bv = BIAS ? bias[col] : 0.0f;
#pragma unroll
      for (int v = 0; v < 8; ++v) {
        const int row = m0 + wm + i * 16 + v + half * 8;
        float val = acc[i][j][v] + bv;
        if (GELU)  val = gelu_tanh(val);
        if (RESID) val += resid[(size_t)row * N + col];
        C[(size_t)row * N + col] = val;
      }
    }
}

// ---------------- write step: softmax attention + EMA slot write ----------------
__global__ __launch_bounds__(256)
void write_step_kernel(const float* __restrict__ x, const float* __restrict__ cs_in,
                       const float* __restrict__ kx, float* __restrict__ cs_out)
{
  const int b = blockIdx.x, tid = threadIdx.x, wave = tid >> 5, lane = tid & 31;
  __shared__ float lg[SS], wts[SS];
  const float* kb  = kx + (size_t)b * DD;
  const float* csb = cs_in + (size_t)b * SS * DD;

#pragma unroll
  for (int si = 0; si < 8; ++si) {
    const int s = wave * 8 + si;
    const float* row = csb + (size_t)s * DD;
    float p = 0.f;
    for (int d = lane; d < DD; d += 32) p += kb[d] * row[d];
#pragma unroll
    for (int off = 16; off; off >>= 1) p += __shfl_down(p, off, 32);
    if (lane == 0) lg[s] = p * 0.03125f;              // D^-0.5
  }
  __syncthreads();
  if (tid < SS) {
    float mx = -1e30f;
    for (int s = 0; s < SS; ++s) mx = fmaxf(mx, lg[s]);
    float sum = 0.f;
    for (int s = 0; s < SS; ++s) sum += __expf(lg[s] - mx);
    wts[tid] = __expf(lg[tid] - mx) / sum;
  }
  __syncthreads();
  const float* xb = x + (size_t)b * DD;
  float* ob = cs_out + (size_t)b * SS * DD;
  for (int i = tid; i < SS * DD; i += 256) {
    const int s = i >> 10, d = i & (DD - 1);
    ob[i] = 0.9f * csb[i] + 0.1f * wts[s] * xb[d];
  }
}

// ---------------- per-domain slot reads -> r256[b, n*64+h] ----------------
__global__ __launch_bounds__(64)
void domain_kernel(const float* __restrict__ x, const float* __restrict__ ds,
                   const float* __restrict__ Wdk, float* __restrict__ r256)
{
  const int nb = blockIdx.x;           // 0..4*B-1
  const int n = nb >> 10, b = nb & (BB - 1);
  const int t = threadIdx.x;           // 0..63
  __shared__ float xs[64], qv[64], lg[64], wv[64];

  xs[t] = x[(size_t)b * DD + n * 64 + t];
  __syncthreads();
  const float* W = Wdk + (size_t)n * 64 * 64;
  float a = 0.f;
  for (int h = 0; h < 64; ++h) a += xs[h] * W[h * 64 + t];
  qv[t] = a;
  __syncthreads();
  const float* dsb = ds + ((size_t)n * BB + b) * 64 * 64;
  float p = 0.f;
  for (int h = 0; h < 64; ++h) p += qv[h] * dsb[t * 64 + h];
  lg[t] = p * 0.125f;                  // HD^-0.5
  __syncthreads();
  float mx = -1e30f;
  for (int s = 0; s < 64; ++s) mx = fmaxf(mx, lg[s]);
  float sum = 0.f;
  for (int s = 0; s < 64; ++s) sum += __expf(lg[s] - mx);
  wv[t] = __expf(lg[t] - mx) / sum;
  __syncthreads();
  float r = 0.f;
  for (int s = 0; s < 64; ++s) r += wv[s] * dsb[s * 64 + t];
  r256[(size_t)b * 256 + n * 64 + t] = r;
}

// ---------------- node attention (ERG / read): logits+softmax+ctx, one pass ----
__global__ __launch_bounds__(256)
void node_attn_kernel(const float* __restrict__ q, const float* __restrict__ cs,
                      const float* __restrict__ extra, float* __restrict__ ctx,
                      int nnodes, float scale)
{
  const int b = blockIdx.x, tid = threadIdx.x, wave = tid >> 5, lane = tid & 31;
  __shared__ float lg[65], wts[65];
  const float* qb  = q + (size_t)b * DD;
  const float* csb = cs + (size_t)b * SS * DD;

  for (int s = wave; s < nnodes; s += 8) {
    const float* row = (s < SS) ? (csb + (size_t)s * DD) : (extra + (size_t)b * DD);
    float p = 0.f;
    for (int d = lane; d < DD; d += 32) p += qb[d] * row[d];
#pragma unroll
    for (int off = 16; off; off >>= 1) p += __shfl_down(p, off, 32);
    if (lane == 0) lg[s] = p * scale;
  }
  __syncthreads();
  if (tid < nnodes) {
    float mx = -1e30f;
    for (int s = 0; s < nnodes; ++s) mx = fmaxf(mx, lg[s]);
    float sum = 0.f;
    for (int s = 0; s < nnodes; ++s) sum += __expf(lg[s] - mx);
    wts[tid] = __expf(lg[tid] - mx) / sum;
  }
  __syncthreads();
  float* ob = ctx + (size_t)b * DD;
  for (int d = tid; d < DD; d += 256) {
    float acc = 0.f;
#pragma unroll 8
    for (int s = 0; s < SS; ++s) acc += wts[s] * csb[(size_t)s * DD + d];
    if (nnodes > SS) acc += wts[SS] * extra[(size_t)b * DD + d];
    ob[d] = acc;
  }
}

// ---------------- LayerNorm ----------------
__global__ __launch_bounds__(256)
void ln_kernel(const float* __restrict__ xin, const float* __restrict__ g,
               const float* __restrict__ bta, float* __restrict__ h)
{
  const int b = blockIdx.x, tid = threadIdx.x, wave = tid >> 5, lane = tid & 31;
  __shared__ float rs[8], rs2[8], mu_s, rstd_s;
  const float* xb = xin + (size_t)b * DD;
  float s = 0.f, s2 = 0.f;
  for (int d = tid; d < DD; d += 256) { float v = xb[d]; s += v; s2 += v * v; }
#pragma unroll
  for (int off = 16; off; off >>= 1) {
    s += __shfl_down(s, off, 32); s2 += __shfl_down(s2, off, 32);
  }
  if (!lane) { rs[wave] = s; rs2[wave] = s2; }
  __syncthreads();
  if (!tid) {
    float S = 0.f, S2 = 0.f;
    for (int w = 0; w < 8; ++w) { S += rs[w]; S2 += rs2[w]; }
    const float mu = S * (1.0f / DD);
    const float var = S2 * (1.0f / DD) - mu * mu;
    mu_s = mu; rstd_s = rsqrtf(var + 1e-5f);
  }
  __syncthreads();
  for (int d = tid; d < DD; d += 256)
    h[(size_t)b * DD + d] = (xb[d] - mu_s) * rstd_s * g[d] + bta[d];
}

// ---------------- launcher ----------------
extern "C" void kernel_launch(void* const* d_in, const int* in_sizes, int n_in,
                              void* d_out, int out_size, void* d_ws, size_t ws_size,
                              hipStream_t stream)
{
  (void)in_sizes; (void)n_in; (void)out_size; (void)ws_size;
  const float* x   = (const float*)d_in[0];
  const float* cs  = (const float*)d_in[1];
  const float* ds  = (const float*)d_in[2];
  const float* Wdk = (const float*)d_in[3];
  const float* Wk  = (const float*)d_in[4];
  const float* eW1 = (const float*)d_in[5];
  const float* eb1 = (const float*)d_in[6];
  const float* eW2 = (const float*)d_in[7];
  const float* eb2 = (const float*)d_in[8];
  const float* dpW = (const float*)d_in[9];
  const float* dpb = (const float*)d_in[10];
  const float* opW = (const float*)d_in[11];
  const float* opb = (const float*)d_in[12];
  const float* lng = (const float*)d_in[13];
  const float* lnb = (const float*)d_in[14];
  const float* mW1 = (const float*)d_in[15];
  const float* mb1 = (const float*)d_in[16];
  const float* mW2 = (const float*)d_in[17];
  const float* mb2 = (const float*)d_in[18];

  float* xout  = (float*)d_out;                  // tuple elem 0: x  [B,D]
  float* csout = xout + (size_t)BB * DD;         // tuple elem 1: conv_state [B,S,D]

  float* w = (float*)d_ws;
  float* kx    = w; w += (size_t)BB * DD;
  float* r256  = w; w += (size_t)BB * 256;
  float* extra = w; w += (size_t)BB * DD;
  float* qbuf  = w; w += (size_t)BB * DD;
  float* ctx   = w; w += (size_t)BB * DD;
  float* x1    = w; w += (size_t)BB * DD;
  float* hbuf  = w; w += (size_t)BB * DD;
  float* tmpF  = w; w += (size_t)BB * FF;
  // bf16 weight cache (converted every launch; deterministic)
  __bf16* bw = (__bf16*)w;
  __bf16* Wkt   = bw; bw += (size_t)DD * DD;   // [N=D][K=D]   (transposed)
  __bf16* dpWb  = bw; bw += (size_t)DD * DD;   // [N=D][K=D]   (already N-major)
  __bf16* opWb  = bw; bw += (size_t)DD * DD;   // [N=D][K=D]
  __bf16* eW1t  = bw; bw += (size_t)DD * FF;   // [N=F][K=D]   (transposed)
  __bf16* eW2t  = bw; bw += (size_t)FF * DD;   // [N=D][K=F]   (transposed)
  __bf16* mW1t  = bw; bw += (size_t)DD * FF;
  __bf16* mW2t  = bw; bw += (size_t)FF * DD;

  const dim3 blk(256);
  const dim3 g8(DD / 128, BB / 128);    // N=1024 GEMMs
  const dim3 g32(FF / 128, BB / 128);   // N=4096 GEMMs

  // 0) weight conversion to bf16 (transpose [K,N]->[N,K] where needed)
  convT_kernel<<<dim3(DD/32, DD/32), blk, 0, stream>>>(Wk,  Wkt,  DD, DD);
  convT_kernel<<<dim3(FF/32, DD/32), blk, 0, stream>>>(eW1, eW1t, DD, FF);
  convT_kernel<<<dim3(DD/32, FF/32), blk, 0, stream>>>(eW2, eW2t, FF, DD);
  convT_kernel<<<dim3(FF/32, DD/32), blk, 0, stream>>>(mW1, mW1t, DD, FF);
  convT_kernel<<<dim3(DD/32, FF/32), blk, 0, stream>>>(mW2, mW2t, FF, DD);
  conv_kernel<<<1024, blk, 0, stream>>>(dpW, dpWb, DD * DD);
  conv_kernel<<<1024, blk, 0, stream>>>(opW, opWb, DD * DD);

  // 1) kx = x @ Wk
  wmma_gemm<0,0,0><<<g8, blk, 0, stream>>>(x, Wkt, nullptr, nullptr, kx, BB, DD, DD, DD);
  // 2) write step -> updated conv_state directly into d_out
  write_step_kernel<<<BB, 256, 0, stream>>>(x, cs, kx, csout);
  // 3) per-domain slot reads
  domain_kernel<<<4 * BB, 64, 0, stream>>>(x, ds, Wdk, r256);
  // 4) extra = r256 @ dp_W[:, :256]^T + dp_b   (K=256, ldb=1024)
  wmma_gemm<0,0,1><<<g8, blk, 0, stream>>>(r256, dpWb, dpb, nullptr, extra, BB, DD, 256, DD);
  // 5) q = x
  hipMemcpyAsync(qbuf, x, (size_t)BB * DD * sizeof(float), hipMemcpyDeviceToDevice, stream);
  // 6) ERG: 4 steps of attention over 65 nodes + FFN residual
  for (int step = 0; step < 4; ++step) {
    node_attn_kernel<<<BB, 256, 0, stream>>>(qbuf, csout, extra, ctx, 65, 0.03125f);
    wmma_gemm<1,0,1><<<g32, blk, 0, stream>>>(ctx, eW1t, eb1, nullptr, tmpF, BB, FF, DD, DD);
    wmma_gemm<0,1,1><<<g8,  blk, 0, stream>>>(tmpF, eW2t, eb2, qbuf, qbuf, BB, DD, FF, FF);
  }
  // 7) read attention over conv slots (read_logit_scale = 1/32)
  node_attn_kernel<<<BB, 256, 0, stream>>>(qbuf, csout, nullptr, ctx, 64, 0.03125f);
  // 8) x1 = x + ctx @ op_W^T + op_b
  wmma_gemm<0,1,1><<<g8, blk, 0, stream>>>(ctx, opWb, opb, x, x1, BB, DD, DD, DD);
  // 9) LayerNorm
  ln_kernel<<<BB, 256, 0, stream>>>(x1, lng, lnb, hbuf);
  // 10) MLP residual -> final x output
  wmma_gemm<1,0,1><<<g32, blk, 0, stream>>>(hbuf, mW1t, mb1, nullptr, tmpF, BB, FF, DD, DD);
  wmma_gemm<0,1,1><<<g8,  blk, 0, stream>>>(tmpF, mW2t, mb2, x1, xout, BB, DD, FF, FF);
}
